// MultiTokenAttention_53369263620412
// MI455X (gfx1250) — matrix operand order repaired and checked by reading, not compiled, not run on hardware
//
#include <hip/hip_runtime.h>

// ---------------- problem constants (from the reference) ----------------
constexpr int Nn  = 2048;   // sequence length
constexpr int Ee  = 1024;   // embed dim
constexpr int Hh  = 16;     // heads
constexpr int Dd  = 64;     // head dim
constexpr int KVd = 1024;   // kv input dim
constexpr int CQ  = 6, CK = 11, CH = 2;   // conv kernel

typedef __bf16 bf16;
typedef __attribute__((ext_vector_type(16))) __bf16 v16bf;
typedef __attribute__((ext_vector_type(8)))  float  v8f;

__device__ __forceinline__ bf16 f2bf(float f) {
  union { float f; unsigned u; } c; c.f = f;
  unsigned r = c.u + 0x7FFFu + ((c.u >> 16) & 1u);   // round-to-nearest-even
  unsigned short h = (unsigned short)(r >> 16);
  return __builtin_bit_cast(bf16, h);
}

// A-operand element e -> K offset (contiguous order: k = 16*s + e, matches B)
__device__ __forceinline__ int akoff(int e, int s) {
  return e + 16 * s;
}

// ---------------------------------------------------------------------------
// Generic GEMM with bias: C = A(MxK) * W(KxNout) + b.
// Each wave owns a 16x64 output strip: one A fragment feeds 4 WMMAs per
// 32-deep k-chunk (4x A-traffic amortization, back-to-back WMMA issue).
// Next W k-slab is prefetched (global_prefetch_b8) while the current one
// is consumed.
// ---------------------------------------------------------------------------
template <bool A_BF16, bool OUT_BF16>
__global__ void wmma_gemm_bias(const void* __restrict__ Av,
                               const float* __restrict__ W,
                               const float* __restrict__ bias,
                               void* __restrict__ Cv,
                               int M, int K, int Nout) {
  const int lane = threadIdx.x & 31;
  const int wave = threadIdx.x >> 5;
  const int m0 = blockIdx.x * 16;
  const int n0 = (blockIdx.y * 8 + wave) * 64;   // 4 consecutive 16-col tiles
  const int m = lane & 15;
  const int s = lane >> 4;
  (void)M;

  v8f acc[4] = {};
  for (int kc = 0; kc < K; kc += 32) {
    v16bf a;
    if (A_BF16) {
      const bf16* ap = (const bf16*)Av + (size_t)(m0 + m) * K + kc;
#pragma unroll
      for (int e = 0; e < 16; ++e) a[e] = ap[akoff(e, s)];
    } else {
      const float* ap = (const float*)Av + (size_t)(m0 + m) * K + kc;
#pragma unroll
      for (int e = 0; e < 16; ++e) a[e] = f2bf(ap[akoff(e, s)]);
    }
    if (kc + 32 < K)  // pull next W slab toward L2 (global_prefetch_b8)
      __builtin_prefetch(&W[(size_t)(kc + 32 + 16 * s) * Nout + n0 + m], 0, 1);
#pragma unroll
    for (int t = 0; t < 4; ++t) {
      v16bf b;
#pragma unroll
      for (int e = 0; e < 16; ++e)
        b[e] = f2bf(W[(size_t)(kc + e + 16 * s) * Nout + n0 + t * 16 + m]);
      acc[t] = __builtin_amdgcn_wmma_f32_16x16x32_bf16(false, a, false, b,
                                                       (short)0, acc[t],
                                                       false, false);
    }
  }
#pragma unroll
  for (int t = 0; t < 4; ++t) {
    const float bv = bias[n0 + t * 16 + m];
#pragma unroll
    for (int r = 0; r < 8; ++r) {
      const int row = m0 + r + 8 * s;
      const float v = acc[t][r] + bv;
      if (OUT_BF16) ((bf16*)Cv)[(size_t)row * Nout + n0 + t * 16 + m] = f2bf(v);
      else          ((float*)Cv)[(size_t)row * Nout + n0 + t * 16 + m] = v;
    }
  }
}

// ---------------------------------------------------------------------------
// Scores: S[h] = (Q_h * K_h^T) / 8, zeroed above the diagonal.
// Each wave owns 4 consecutive k-tiles of one q-tile: A fragments (2 chunks
// of the 64-deep head dim) are loaded once and feed up to 8 WMMAs.
// Fully masked tiles are written as zeros without compute.
// ---------------------------------------------------------------------------
__global__ void wmma_scores(const bf16* __restrict__ Qb,
                            const bf16* __restrict__ Kb,
                            float* __restrict__ S) {
  const int qt = blockIdx.x, kts = blockIdx.y * 4, h = blockIdx.z;
  const int lane = threadIdx.x & 31;
  const int m = lane & 15, s = lane >> 4;
  const int q0 = qt * 16;
  float* Sh = S + (size_t)h * Nn * Nn;

  v16bf a0, a1;
  if (kts <= qt) {  // at least one live tile: load A fragments once
    const bf16* ap = Qb + (size_t)(q0 + m) * Ee + h * Dd;
#pragma unroll
    for (int e = 0; e < 16; ++e) a0[e] = ap[akoff(e, s)];
#pragma unroll
    for (int e = 0; e < 16; ++e) a1[e] = ap[32 + akoff(e, s)];
  }

#pragma unroll
  for (int t = 0; t < 4; ++t) {
    const int kt = kts + t;
    const int k0 = kt * 16;
    if (kt > qt) {  // fully masked tile -> zeros
#pragma unroll
      for (int r = 0; r < 8; ++r)
        Sh[(size_t)(q0 + r + 8 * s) * Nn + k0 + m] = 0.f;
      continue;
    }
    v8f acc = {};
    // B[kk][n] = K[k0+n][d0+kk] -> contiguous per lane
    const bf16* bp = Kb + (size_t)(k0 + m) * Ee + h * Dd + 16 * s;
    v16bf b;
#pragma unroll
    for (int e = 0; e < 16; ++e) b[e] = bp[e];
    acc = __builtin_amdgcn_wmma_f32_16x16x32_bf16(false, a0, false, b,
                                                  (short)0, acc, false, false);
#pragma unroll
    for (int e = 0; e < 16; ++e) b[e] = bp[32 + e];
    acc = __builtin_amdgcn_wmma_f32_16x16x32_bf16(false, a1, false, b,
                                                  (short)0, acc, false, false);
#pragma unroll
    for (int r = 0; r < 8; ++r) {
      const int row = q0 + r + 8 * s, col = k0 + m;
      Sh[(size_t)row * Nn + col] = (col <= row) ? acc[r] * 0.125f : 0.f;
    }
  }
}

// ---------------------------------------------------------------------------
// Grouped conv (2x6x11, groups=8) + causal mask + row softmax -> bf16 attn.
// One block per (q,h): stages the 12 needed score rows through LDS in
// 512-column chunks with a 5-wide halo, conv in registers, two-stage
// block reductions for max and sum.
// ---------------------------------------------------------------------------
__global__ void conv_softmax(const float* __restrict__ S,
                             const float* __restrict__ cw,
                             bf16* __restrict__ attn) {
  const int q = blockIdx.x, h = blockIdx.y;
  const int tid = threadIdx.x;
  const int g0 = h & ~1;            // first input head of this group

  __shared__ float swt[CH * CQ * CK];   // 132 weights
  __shared__ float sm[12][528];         // 12 rows x (512 + 10) cols (+pad)
  __shared__ float red[8];
  __shared__ float bmax, bsum;

  if (tid < CH * CQ * CK) swt[tid] = cw[h * (CH * CQ * CK) + tid];

  float cv[8];
  float rmax = -3.0e38f;

  for (int c = 0; c < 4; ++c) {           // four 512-wide column chunks
    __syncthreads();
    for (int idx = tid; idx < 12 * 522; idx += 256) {
      const int r = idx / 522, x = idx - r * 522;
      const int ci = r / CQ, iq = r - ci * CQ;
      const int row = q - (CQ - 1) + iq;
      const int col = c * 512 + x - 5;
      float v = 0.f;
      if (row >= 0 && col >= 0 && col < Nn)
        v = S[((size_t)(g0 + ci) * Nn + row) * Nn + col];
      sm[r][x] = v;
    }
    __syncthreads();
#pragma unroll
    for (int j = 0; j < 2; ++j) {
      const int k = c * 512 + j * 256 + tid;
      float acc = 0.f;
#pragma unroll
      for (int ci = 0; ci < CH; ++ci)
#pragma unroll
        for (int iq = 0; iq < CQ; ++iq) {
          const float* sr = &sm[ci * CQ + iq][j * 256 + tid];
          const float* wr = &swt[ci * (CQ * CK) + iq * CK];
#pragma unroll
          for (int ik = 0; ik < CK; ++ik) acc = fmaf(wr[ik], sr[ik], acc);
        }
      cv[c * 2 + j] = acc;
      if (k <= q) rmax = fmaxf(rmax, acc);
    }
  }

  // block max
#pragma unroll
  for (int o = 16; o > 0; o >>= 1) rmax = fmaxf(rmax, __shfl_xor(rmax, o, 32));
  if ((tid & 31) == 0) red[tid >> 5] = rmax;
  __syncthreads();
  if (tid == 0) {
    float mx = red[0];
#pragma unroll
    for (int w = 1; w < 8; ++w) mx = fmaxf(mx, red[w]);
    bmax = mx;
  }
  __syncthreads();
  const float mx = bmax;

  // exp + block sum
  float lsum = 0.f;
#pragma unroll
  for (int i = 0; i < 8; ++i) {
    const int k = (i >> 1) * 512 + (i & 1) * 256 + tid;
    const float e = (k <= q) ? __expf(cv[i] - mx) : 0.f;
    cv[i] = e;
    lsum += e;
  }
#pragma unroll
  for (int o = 16; o > 0; o >>= 1) lsum += __shfl_xor(lsum, o, 32);
  if ((tid & 31) == 0) red[tid >> 5] = lsum;
  __syncthreads();
  if (tid == 0) {
    float sm2 = 0.f;
#pragma unroll
    for (int w = 0; w < 8; ++w) sm2 += red[w];
    bsum = sm2;
  }
  __syncthreads();
  const float inv = 1.0f / bsum;

  bf16* arow = attn + ((size_t)h * Nn + q) * Nn;
#pragma unroll
  for (int i = 0; i < 8; ++i) {
    const int k = (i >> 1) * 512 + (i & 1) * 256 + tid;
    arow[k] = f2bf(cv[i] * inv);
  }
}

// ---------------------------------------------------------------------------
// ctx_h = attn_h (2048x2048 bf16) * V_h (2048x64 bf16); each wave owns a
// full 16x64 (whole head) strip: one attn A-fragment per k-chunk feeds 4
// WMMAs. K-loop is bounded by causality (attn row q is zero past column q).
// ---------------------------------------------------------------------------
__global__ void wmma_ctx(const bf16* __restrict__ attn,
                         const bf16* __restrict__ Vb,
                         bf16* __restrict__ Ctx) {
  const int qt = blockIdx.x, h = blockIdx.y;
  const int lane = threadIdx.x & 31;
  const int m = lane & 15, s = lane >> 4;
  const int q0 = qt * 16;
  const bf16* Ah = attn + (size_t)h * Nn * Nn;
  const int kend = ((q0 + 16 + 31) >> 5) << 5;   // causal extent, mult of 32

  v8f acc[4] = {};
  for (int kc = 0; kc < kend; kc += 32) {
    v16bf a;
    const bf16* ap = Ah + (size_t)(q0 + m) * Nn + kc;
#pragma unroll
    for (int e = 0; e < 16; ++e) a[e] = ap[akoff(e, s)];
#pragma unroll
    for (int t = 0; t < 4; ++t) {
      v16bf b;
#pragma unroll
      for (int e = 0; e < 16; ++e)
        b[e] = Vb[(size_t)(kc + e + 16 * s) * Ee + h * Dd + t * 16 + m];
      acc[t] = __builtin_amdgcn_wmma_f32_16x16x32_bf16(false, a, false, b,
                                                       (short)0, acc[t],
                                                       false, false);
    }
  }
#pragma unroll
  for (int t = 0; t < 4; ++t)
#pragma unroll
    for (int r = 0; r < 8; ++r)
      Ctx[(size_t)(q0 + r + 8 * s) * Ee + h * Dd + t * 16 + m] = f2bf(acc[t][r]);
}

// ---------------------------------------------------------------------------
extern "C" void kernel_launch(void* const* d_in, const int* in_sizes, int n_in,
                              void* d_out, int out_size, void* d_ws, size_t ws_size,
                              hipStream_t stream) {
  (void)in_sizes; (void)n_in; (void)out_size; (void)ws_size;
  const float* query = (const float*)d_in[0];
  const float* key   = (const float*)d_in[1];
  const float* value = (const float*)d_in[2];
  const float* Wq    = (const float*)d_in[3];
  const float* bq    = (const float*)d_in[4];
  const float* Wk    = (const float*)d_in[5];
  const float* bk    = (const float*)d_in[6];
  const float* Wv    = (const float*)d_in[7];
  const float* bv    = (const float*)d_in[8];
  const float* cw    = (const float*)d_in[9];
  const float* Wo    = (const float*)d_in[10];
  const float* bo    = (const float*)d_in[11];

  char* p = (char*)d_ws;
  auto take = [&](size_t bytes) {
    char* r = p;
    p += (bytes + 255) & ~(size_t)255;
    return r;
  };
  bf16*  Qb = (bf16*) take((size_t)Nn * Ee * sizeof(bf16));
  bf16*  Kb = (bf16*) take((size_t)Nn * Ee * sizeof(bf16));
  bf16*  Vb = (bf16*) take((size_t)Nn * Ee * sizeof(bf16));
  float* S  = (float*)take((size_t)Hh * Nn * Nn * sizeof(float));
  bf16*  At = (bf16*) take((size_t)Hh * Nn * Nn * sizeof(bf16));
  bf16*  Cx = (bf16*) take((size_t)Nn * Ee * sizeof(bf16));

  const dim3 gemmGrid(Nn / 16, Ee / 64 / 8);   // (128, 2): 16x64 strip / wave
  wmma_gemm_bias<false, true><<<gemmGrid, 256, 0, stream>>>(query, Wq, bq, Qb, Nn, Ee,  Ee);
  wmma_gemm_bias<false, true><<<gemmGrid, 256, 0, stream>>>(key,   Wk, bk, Kb, Nn, KVd, Ee);
  wmma_gemm_bias<false, true><<<gemmGrid, 256, 0, stream>>>(value, Wv, bv, Vb, Nn, KVd, Ee);

  wmma_scores<<<dim3(Nn / 16, Nn / 64, Hh), 32, 0, stream>>>(Qb, Kb, S);

  conv_softmax<<<dim3(Nn, Hh), 256, 0, stream>>>(S, cw, At);

  wmma_ctx<<<dim3(Nn / 16, Hh), 32, 0, stream>>>(At, Vb, Cx);

  wmma_gemm_bias<true, false><<<gemmGrid, 256, 0, stream>>>(Cx, Wo, bo, (float*)d_out, Nn, Ee, Ee);
}
